// MambaBlock_48069273977388
// MI455X (gfx1250) — matrix operand order repaired and checked by reading, not compiled, not run on hardware
//
#include <hip/hip_runtime.h>
#include <hip/hip_bf16.h>
#include <math.h>

// Problem dims (fixed by the reference)
#define B__     2
#define L__     2048
#define DMODEL  1024
#define DINNER  2048
#define DSTATE  16
#define DCONV   4
#define DTRANK  64
#define NFEAT   (DTRANK + 2 * DSTATE)   // 96
#define ROWS    (B__ * L__)             // 4096 rows for all GEMMs

typedef __attribute__((ext_vector_type(16))) _Float16 v16h;
typedef __attribute__((ext_vector_type(8)))  _Float16 v8h;
typedef __attribute__((ext_vector_type(8)))  float    v8f;

// Fragment loaders (CDNA5 ISA 7.12.2 layouts; 16B-aligned b128 chunks).
//   A (16x32, lane=M): lanes 0-15 hold K {0..7,16..23}; lanes 16-31 {8..15,24..31}
//   B (32x16, lane=N): lanes 0-15 hold K {0..7,16..23}; lanes 16-31 {8..15,24..31}
__device__ __forceinline__ v16h load_frag_a(const _Float16* __restrict__ pa, int k0) {
  const v8h a0 = *(const v8h*)(pa + k0);
  const v8h a1 = *(const v8h*)(pa + k0 + 16);
  return __builtin_shufflevector(a0, a1,
      0, 1, 2, 3, 4, 5, 6, 7, 8, 9, 10, 11, 12, 13, 14, 15);
}
__device__ __forceinline__ v16h load_frag_b(const _Float16* __restrict__ pb, int k0) {
  const v8h b0 = *(const v8h*)(pb + k0);
  const v8h b1 = *(const v8h*)(pb + k0 + 16);
  return __builtin_shufflevector(b0, b1,
      0, 1, 2, 3, 4, 5, 6, 7, 8, 9, 10, 11, 12, 13, 14, 15);
}

// ---------------------------------------------------------------------------
// Generic C[M,N] = A[M,K] * W[N,K]^T,  f16 operands, f32 accumulate/output.
// Block = 8 waves; each wave owns a 16(M) x NSUB*16(N) tile.
// Grid: x = N/(NSUB*16) exactly, y = M/128 exactly -> no bounds checks.
// K must be a multiple of 64 (holds for all launches: 1024, 2048, 64).
// Ping-pong software pipeline (manual unroll x2): each WMMA chain consumes the
// buffer loaded one half-iteration earlier; no buffer rotation copies.
// ---------------------------------------------------------------------------
template <int NSUB>
__global__ __launch_bounds__(256)
void gemm_xwT_wmma(const _Float16* __restrict__ A, const _Float16* __restrict__ W,
                   float* __restrict__ C, int K, int lda, int ldw, int ldc) {
  const int lane = threadIdx.x & 31;
  const int wave = threadIdx.x >> 5;
  const int row0 = (blockIdx.y * 8 + wave) * 16;
  const int colBase = blockIdx.x * (NSUB * 16);

  const int r   = lane & 15;             // A row within tile
  const int kbA = (lane >> 4) * 8;       // A K-phase
  const int n   = lane & 15;             // B col within subtile
  const int kbB = (lane >> 4) * 8;       // B K-phase

  const _Float16* pa = A + (size_t)(row0 + r) * lda + kbA;
  const _Float16* pb[NSUB];
#pragma unroll
  for (int s = 0; s < NSUB; ++s)
    pb[s] = W + (size_t)(colBase + s * 16 + n) * ldw + kbB;

  v8f acc[NSUB];
#pragma unroll
  for (int s = 0; s < NSUB; ++s)
#pragma unroll
    for (int j = 0; j < 8; ++j) acc[s][j] = 0.0f;

  // Prologue: slab 0 -> buffer 0
  v16h a0f = load_frag_a(pa, 0);
  v16h b0f[NSUB];
#pragma unroll
  for (int s = 0; s < NSUB; ++s) b0f[s] = load_frag_b(pb[s], 0);

  v16h a1f;
  v16h b1f[NSUB];

  int k0 = 0;
  for (; k0 + 64 < K; k0 += 64) {
    // load slab k0+32 -> buffer 1, then consume buffer 0
    a1f = load_frag_a(pa, k0 + 32);
#pragma unroll
    for (int s = 0; s < NSUB; ++s) b1f[s] = load_frag_b(pb[s], k0 + 32);
    __builtin_prefetch(pa + k0 + 64, 0, 1);   // L2 prefetch (global_prefetch_b8)
#pragma unroll
    for (int s = 0; s < NSUB; ++s)
      acc[s] = __builtin_amdgcn_wmma_f32_16x16x32_f16(
          false, a0f, false, b0f[s], (short)0, acc[s], false, false);

    // load slab k0+64 -> buffer 0, then consume buffer 1
    a0f = load_frag_a(pa, k0 + 64);
#pragma unroll
    for (int s = 0; s < NSUB; ++s) b0f[s] = load_frag_b(pb[s], k0 + 64);
#pragma unroll
    for (int s = 0; s < NSUB; ++s)
      acc[s] = __builtin_amdgcn_wmma_f32_16x16x32_f16(
          false, a1f, false, b1f[s], (short)0, acc[s], false, false);
  }

  // Tail: exactly two slabs remain (K % 64 == 0)
  a1f = load_frag_a(pa, k0 + 32);
#pragma unroll
  for (int s = 0; s < NSUB; ++s) b1f[s] = load_frag_b(pb[s], k0 + 32);
#pragma unroll
  for (int s = 0; s < NSUB; ++s)
    acc[s] = __builtin_amdgcn_wmma_f32_16x16x32_f16(
        false, a0f, false, b0f[s], (short)0, acc[s], false, false);
#pragma unroll
  for (int s = 0; s < NSUB; ++s)
    acc[s] = __builtin_amdgcn_wmma_f32_16x16x32_f16(
        false, a1f, false, b1f[s], (short)0, acc[s], false, false);

  // D layout: VGPR j, lanes 0-15 -> M=j, lanes 16-31 -> M=j+8; N = lane&15
  const int rofs = (lane >> 4) * 8;
#pragma unroll
  for (int s = 0; s < NSUB; ++s) {
    const int c0 = colBase + s * 16;
#pragma unroll
    for (int j = 0; j < 8; ++j)
      C[(size_t)(row0 + rofs + j) * ldc + (c0 + n)] = acc[s][j];
  }
}

// ---------------------------------------------------------------------------
// Elementwise fp32 -> fp16 (n must be a multiple of 4; all our sizes are)
// ---------------------------------------------------------------------------
__global__ __launch_bounds__(256)
void f32_to_f16_kernel(const float* __restrict__ in, _Float16* __restrict__ out,
                       size_t n4) {
  const size_t i = ((size_t)blockIdx.x * blockDim.x + threadIdx.x);
  if (i >= n4) return;
  const float4 v = *(const float4*)(in + i * 4);
  _Float16* o = out + i * 4;
  o[0] = (_Float16)v.x; o[1] = (_Float16)v.y;
  o[2] = (_Float16)v.z; o[3] = (_Float16)v.w;
}

// ---------------------------------------------------------------------------
// Extract dt_lr = dbc[:, 0:64] as f16 (A-matrix for dt_proj GEMM)
// ---------------------------------------------------------------------------
__global__ __launch_bounds__(256)
void extract_dtlr_kernel(const float* __restrict__ dbc, _Float16* __restrict__ dtlr) {
  const size_t idx = (size_t)blockIdx.x * blockDim.x + threadIdx.x;  // ROWS*64
  const size_t row = idx >> 6;
  const int    c   = (int)(idx & 63);
  dtlr[idx] = (_Float16)dbc[row * NFEAT + c];
}

// ---------------------------------------------------------------------------
// Causal depthwise conv (width 4) + bias + SiLU.
// Dual output: f32 (for the scan) and f16 (A-matrix for x_proj GEMM).
// ---------------------------------------------------------------------------
__global__ __launch_bounds__(256)
void conv_silu_kernel(const float* __restrict__ xz, const float* __restrict__ cw,
                      const float* __restrict__ cb, float* __restrict__ xc,
                      _Float16* __restrict__ xc16) {
  const size_t idx = (size_t)blockIdx.x * blockDim.x + threadIdx.x;
  if (idx >= (size_t)ROWS * DINNER) return;
  const int    e   = (int)(idx % DINNER);
  const size_t row = idx / DINNER;       // b*L + l
  const int    l   = (int)(row % L__);
  float s = cb[e];
#pragma unroll
  for (int k = 0; k < DCONV; ++k) {
    const int ll = l - (DCONV - 1) + k;
    if (ll >= 0)
      s += xz[(row + (size_t)(ll - l)) * (2 * DINNER) + e] * cw[e * DCONV + k];
  }
  s = s / (1.0f + __expf(-s));           // SiLU
  xc[row * DINNER + e]   = s;
  xc16[row * DINNER + e] = (_Float16)s;
}

// ---------------------------------------------------------------------------
// Selective scan, fully fused:
//   dt = softplus(dtpre + dt_bias); dA = exp(dt*A); h = dA*h + dt*B*x;
//   y = (h.C + D*x) * silu(z)   -> written as f16 (only consumer is out_proj)
// One lane per (batch, channel); 16 states in registers; B/C staged in LDS.
// ---------------------------------------------------------------------------
__global__ __launch_bounds__(256)
void scan_kernel(const float* __restrict__ dbc, const float* __restrict__ dtpre,
                 const float* __restrict__ dtb, const float* __restrict__ A_log,
                 const float* __restrict__ Dp,  const float* __restrict__ xc,
                 const float* __restrict__ xz,  _Float16* __restrict__ y16) {
  const int b = blockIdx.x >> 3;                       // batch
  const int d = ((blockIdx.x & 7) << 8) + threadIdx.x; // channel 0..2047

  float Areg[DSTATE], h[DSTATE];
#pragma unroll
  for (int nn = 0; nn < DSTATE; ++nn) {
    Areg[nn] = -expf(A_log[d * DSTATE + nn]);
    h[nn] = 0.0f;
  }
  const float bias = dtb[d];
  const float Dd   = Dp[d];

  __shared__ float sBC[32];  // [0:16) = B_t, [16:32) = C_t
  for (int l = 0; l < L__; ++l) {
    const size_t row = (size_t)b * L__ + l;
    __syncthreads();  // protect sBC from previous iteration's readers
    if (threadIdx.x < 32)
      sBC[threadIdx.x] = dbc[row * NFEAT + DTRANK + threadIdx.x];
    __syncthreads();

    float dt = dtpre[row * DINNER + d] + bias;
    dt = (dt > 20.0f) ? dt : log1pf(expf(dt));         // softplus
    const float xv  = xc[row * DINNER + d];
    const float dtx = dt * xv;

    float yacc = 0.0f;
#pragma unroll
    for (int nn = 0; nn < DSTATE; ++nn) {
      const float dA = expf(dt * Areg[nn]);
      h[nn] = dA * h[nn] + dtx * sBC[nn];
      yacc += h[nn] * sBC[16 + nn];
    }

    const float zv = xz[row * (2 * DINNER) + DINNER + d];
    const float sz = zv / (1.0f + __expf(-zv));        // silu(z)
    y16[row * DINNER + d] = (_Float16)((yacc + Dd * xv) * sz);
  }
}

// ---------------------------------------------------------------------------
// LayerNorm over DMODEL=1024: one 256-thread block per row, LDS tree reduce.
// ---------------------------------------------------------------------------
__global__ __launch_bounds__(256)
void layernorm_kernel(const float* __restrict__ in, const float* __restrict__ gamma,
                      const float* __restrict__ beta, float* __restrict__ out) {
  __shared__ float s_sum[256];
  __shared__ float s_sq[256];
  const size_t base = (size_t)blockIdx.x * DMODEL;
  float v[4];
  float sum = 0.0f, sq = 0.0f;
#pragma unroll
  for (int i = 0; i < 4; ++i) {
    v[i] = in[base + threadIdx.x + i * 256];
    sum += v[i];
    sq  += v[i] * v[i];
  }
  s_sum[threadIdx.x] = sum;
  s_sq[threadIdx.x]  = sq;
  __syncthreads();
  for (int off = 128; off > 0; off >>= 1) {
    if ((int)threadIdx.x < off) {
      s_sum[threadIdx.x] += s_sum[threadIdx.x + off];
      s_sq[threadIdx.x]  += s_sq[threadIdx.x + off];
    }
    __syncthreads();
  }
  const float mu   = s_sum[0] * (1.0f / DMODEL);
  const float var  = s_sq[0] * (1.0f / DMODEL) - mu * mu;
  const float rstd = rsqrtf(var + 1e-5f);
#pragma unroll
  for (int i = 0; i < 4; ++i) {
    const int c = threadIdx.x + i * 256;
    out[base + c] = (v[i] - mu) * rstd * gamma[c] + beta[c];
  }
}

// ---------------------------------------------------------------------------
extern "C" void kernel_launch(void* const* d_in, const int* in_sizes, int n_in,
                              void* d_out, int out_size, void* d_ws, size_t ws_size,
                              hipStream_t stream) {
  const float* x        = (const float*)d_in[0];   // [B,L,DMODEL]
  const float* in_w     = (const float*)d_in[1];   // [2*DINNER, DMODEL]
  const float* conv_w   = (const float*)d_in[2];   // [DINNER, DCONV]
  const float* conv_b   = (const float*)d_in[3];   // [DINNER]
  const float* xproj_w  = (const float*)d_in[4];   // [NFEAT, DINNER]
  const float* dtproj_w = (const float*)d_in[5];   // [DINNER, DTRANK]
  const float* dtproj_b = (const float*)d_in[6];   // [DINNER]
  const float* A_log    = (const float*)d_in[7];   // [DINNER, DSTATE]
  const float* Dp       = (const float*)d_in[8];   // [DINNER]
  const float* out_w    = (const float*)d_in[9];   // [DMODEL, DINNER]
  const float* g        = (const float*)d_in[10];  // [DMODEL]
  const float* bta      = (const float*)d_in[11];  // [DMODEL]
  float* out = (float*)d_out;

  // ---------------- workspace layout ----------------
  // f32 region
  float* xz    = (float*)d_ws;                        // ROWS * 4096
  float* xc    = xz    + (size_t)ROWS * (2 * DINNER); // ROWS * 2048
  float* dtpre = xc    + (size_t)ROWS * DINNER;       // ROWS * 2048
  float* dbc   = dtpre + (size_t)ROWS * DINNER;       // ROWS * 96
  float* f32_end = dbc + (size_t)ROWS * NFEAT;
  float* obuf  = xz;                                  // ROWS*1024 alias (xz dead)
  // f16 region
  _Float16* x16    = (_Float16*)f32_end;
  _Float16* w_in   = x16    + (size_t)ROWS * DMODEL;        // 2*DINNER x DMODEL
  _Float16* w_xp   = w_in   + (size_t)(2 * DINNER) * DMODEL;
  _Float16* w_dt   = w_xp   + (size_t)NFEAT * DINNER;
  _Float16* w_out  = w_dt   + (size_t)DINNER * DTRANK;
  _Float16* xc16   = w_out  + (size_t)DMODEL * DINNER;
  _Float16* dtlr16 = xc16   + (size_t)ROWS * DINNER;
  _Float16* y16    = dtlr16 + (size_t)ROWS * DTRANK;

  const dim3 blk(256);
  auto cvt = [&](const float* src, _Float16* dst, size_t n) {
    f32_to_f16_kernel<<<(unsigned)(n / 4 / 256), blk, 0, stream>>>(src, dst, n / 4);
  };

  // 0) one-time f32 -> f16 conversions (inputs + weights)
  cvt(x,        x16,   (size_t)ROWS * DMODEL);
  cvt(in_w,     w_in,  (size_t)(2 * DINNER) * DMODEL);
  cvt(xproj_w,  w_xp,  (size_t)NFEAT * DINNER);
  cvt(dtproj_w, w_dt,  (size_t)DINNER * DTRANK);
  cvt(out_w,    w_out, (size_t)DMODEL * DINNER);

  // 1) in_proj: xz = x * in_w^T          [4096 x 4096, K=1024]
  gemm_xwT_wmma<4><<<dim3((2 * DINNER) / 64, ROWS / 128), blk, 0, stream>>>(
      x16, w_in, xz, DMODEL, DMODEL, DMODEL, 2 * DINNER);

  // 2) depthwise causal conv + SiLU -> xc (f32) + xc16 (f16)
  conv_silu_kernel<<<(unsigned)(((size_t)ROWS * DINNER) / 256), blk, 0, stream>>>(
      xz, conv_w, conv_b, xc, xc16);

  // 3) x_proj: dbc = xc * xproj_w^T      [4096 x 96, K=2048]  (NSUB=2 -> 3 col-blocks)
  gemm_xwT_wmma<2><<<dim3(NFEAT / 32, ROWS / 128), blk, 0, stream>>>(
      xc16, w_xp, dbc, DINNER, DINNER, DINNER, NFEAT);

  // 4) dt_lr slice -> f16
  extract_dtlr_kernel<<<(unsigned)(((size_t)ROWS * DTRANK) / 256), blk, 0, stream>>>(
      dbc, dtlr16);

  // 5) dt_proj: dtpre = dt_lr * dtproj_w^T   [4096 x 2048, K=64]
  gemm_xwT_wmma<4><<<dim3(DINNER / 64, ROWS / 128), blk, 0, stream>>>(
      dtlr16, w_dt, dtpre, DTRANK, DTRANK, DTRANK, DINNER);

  // 6) fused selective scan (+softplus, +D-skip, +silu(z) gate) -> y16
  scan_kernel<<<16, blk, 0, stream>>>(dbc, dtpre, dtproj_b, A_log, Dp, xc, xz, y16);

  // 7) out_proj: obuf = y * out_w^T      [4096 x 1024, K=2048]
  gemm_xwT_wmma<4><<<dim3(DMODEL / 64, ROWS / 128), blk, 0, stream>>>(
      y16, w_out, obuf, DINNER, DINNER, DINNER, DMODEL);

  // 8) LayerNorm -> d_out
  layernorm_kernel<<<ROWS, blk, 0, stream>>>(obuf, g, bta, out);
}
